// Conv2DKAN_50611894616762
// MI455X (gfx1250) — compile-verified, matrix operand order checked by reading, not run on hardware
//
#include <hip/hip_runtime.h>

typedef __attribute__((ext_vector_type(16))) _Float16 v16h;
typedef __attribute__((ext_vector_type(8)))  _Float16 v8h;
typedef __attribute__((ext_vector_type(8)))  float    v8f;

#define IN_SIZE  288                   // 3*3*32
#define NB       9                     // 8 spline basis + silu
#define KTOT     (IN_SIZE * NB)        // 2592
#define NCHUNK   (KTOT / 32)           // 81 K-chunks of 32
#define FOUT     128
#define NPATCH   28800                 // 32*30*30
#define MTILE    64
#define WFRAG_U32   (NCHUNK * 8 * 32 * 8)    // 165888 dwords
#define WFRAG_BYTES (WFRAG_U32 * 4)          // 663552 B
#define BIAS_BYTES  512
#define PLANE_ELEMS (32 * 32 * 32 * 32)      // 1,048,576 per basis plane
#define PLANE_BYTES (PLANE_ELEMS * 2)        // 2 MB (f16)
#define PLANES_OFF  (WFRAG_BYTES + BIAS_BYTES)
#define ABUF_BYTES  (4 * NB * 32 * 32)       // 36,864 B per A-stage buffer

// ---------------------------------------------------------------------------
// Cox-de Boor order 3 on uniform knots t_j = -2.2 + 0.4*j (j=0..11).
// ---------------------------------------------------------------------------
__device__ __forceinline__ void spline9(float xv, float* v9) {
  const float h = 0.4f, t0 = -2.2f;
  const float inv1 = 1.0f / h, inv2 = 1.0f / (2.0f * h), inv3 = 1.0f / (3.0f * h);
  float b0[11], b1[10], b2[9];
#pragma unroll
  for (int q = 0; q < 11; ++q) {
    float tj = t0 + h * q;
    b0[q] = (xv >= tj && xv < tj + h) ? 1.0f : 0.0f;
  }
#pragma unroll
  for (int q = 0; q < 10; ++q)
    b1[q] = ((xv - (t0 + h * q)) * b0[q] + ((t0 + h * (q + 2)) - xv) * b0[q + 1]) * inv1;
#pragma unroll
  for (int q = 0; q < 9; ++q)
    b2[q] = ((xv - (t0 + h * q)) * b1[q] + ((t0 + h * (q + 3)) - xv) * b1[q + 1]) * inv2;
#pragma unroll
  for (int q = 0; q < 8; ++q)
    v9[q] = ((xv - (t0 + h * q)) * b2[q] + ((t0 + h * (q + 4)) - xv) * b2[q + 1]) * inv3;
  v9[8] = xv / (1.0f + __expf(-xv));        // silu as 9th basis fn
}

// ---------------------------------------------------------------------------
// Phase 1a: pack augmented weights into WMMA B-fragment register order.
// ---------------------------------------------------------------------------
__global__ __launch_bounds__(256) void kan_prep(
    const float* __restrict__ sk, const float* __restrict__ sc,
    const float* __restrict__ kb, const float* __restrict__ cb,
    unsigned int* __restrict__ wfrag, float* __restrict__ biasout) {
  int gid = blockIdx.x * 256 + threadIdx.x;
  int d    = gid & 7;
  int L    = (gid >> 3) & 31;
  int nt   = (gid >> 8) & 7;
  int c    = gid >> 11;
  int half = L >> 4;
  int col  = nt * 16 + (L & 15);
  int j    = (d < 4) ? (d * 2 + half * 8) : (16 + (d - 4) * 2 + half * 8);

  unsigned int packed = 0;
#pragma unroll
  for (int e = 0; e < 2; ++e) {
    int k = c * 32 + j + e;                 // K = b*288 + i (basis-major)
    int b = k / IN_SIZE;
    int i = k - b * IN_SIZE;
    float s = sc[i * FOUT + col];
    float v = (b < 8) ? sk[(i * 8 + b) * FOUT + col] * s : s;
    _Float16 hv = (_Float16)v;
    packed |= ((unsigned int)__builtin_bit_cast(unsigned short, hv)) << (16 * e);
  }
  wfrag[gid] = packed;
  if (gid < FOUT) biasout[gid] = kb[gid] + cb[gid];
}

// ---------------------------------------------------------------------------
// Phase 1b: basis planes, 8 elements/thread -> b128 plane stores.
// ---------------------------------------------------------------------------
__global__ __launch_bounds__(256) void kan_basis(
    const float* __restrict__ x, _Float16* __restrict__ planes) {
  int g8 = (blockIdx.x * 256 + threadIdx.x) * 8;
  float v9[8][9];
#pragma unroll
  for (int e = 0; e < 8; ++e) spline9(x[g8 + e], v9[e]);
#pragma unroll
  for (int b = 0; b < NB; ++b) {
    v8h pk;
#pragma unroll
    for (int e = 0; e < 8; ++e) pk[e] = (_Float16)v9[e][b];
    *(v8h*)(planes + (size_t)b * PLANE_ELEMS + g8) = pk;
  }
}

// ---------------------------------------------------------------------------
// Phase 2: WMMA GEMM. Double-buffered async A staging + software-pipelined
// K-loop so B loads for step b+1 fly under the WMMAs of step b.
// ---------------------------------------------------------------------------
__global__ __launch_bounds__(256) void kan_gemm(
    const v16h* __restrict__ wfrag,
    const _Float16* __restrict__ planes,
    const float* __restrict__ bias,
    float* __restrict__ out) {
  extern __shared__ char smem[];          // 2 * ABUF_BYTES, dynamic

  const int tid  = threadIdx.x;
  const int lane = tid & 31;
  const int wave = tid >> 5;
  const int mt   = wave & 3;              // consumer: 16-row m-tile
  const int wn   = wave >> 2;             // consumer: 64-col half
  const int tile_base = blockIdx.x * MTILE;

  // stager mapping: wave w stages patches w*8..w*8+7; lane = (patch, 16B chunk)
  const int pm   = wave * 8 + (lane >> 2);
  const int c4   = lane & 3;
  const int LS   = (pm & 15) + ((c4 & 1) << 4);
  const int mtS  = pm >> 4;
  const int segS = c4 >> 1;
  int pgS = tile_base + pm;
  int woS = pgS % 30; int t1 = pgS / 30;
  int hoS = t1 % 30;  int biS = t1 / 30;

  const unsigned ldsBase = (unsigned)(size_t)(void*)smem;
  const unsigned fragOff = (unsigned)(mtS * NB * 32 + LS) * 32u + (unsigned)(segS * 16);
  const unsigned long long gBase = (unsigned long long)(size_t)planes;

  // issue the 9 async fragment loads (one per basis plane) for slab s2
  auto stage = [&](int s2) {
    int kh = s2 / 3, kw = s2 - 3 * (s2 / 3);
    unsigned long long g = gBase +
        (unsigned long long)((((biS * 32 + hoS + kh) * 32) + (woS + kw)) * 64 + c4 * 16);
    unsigned lb = ldsBase + (unsigned)(s2 & 1) * (unsigned)ABUF_BYTES + fragOff;
#pragma unroll
    for (int b = 0; b < NB; ++b) {
      unsigned lds_addr = lb + (unsigned)b * 1024u;
      unsigned long long gaddr = g + (unsigned long long)b * PLANE_BYTES;
      asm volatile("global_load_async_to_lds_b128 %0, %1, off"
                   :: "v"(lds_addr), "v"(gaddr) : "memory");
    }
  };

  v8f acc[4] = {};
  stage(0);                                        // prologue

  for (int s = 0; s < 9; ++s) {
    if (s < 8) stage(s + 1);                       // overlap next slab's staging
    // loads complete in order: <=9 outstanding ==> slab-s group has landed
    if (s < 8) asm volatile("s_wait_asynccnt 0x9" ::: "memory");
    else       asm volatile("s_wait_asynccnt 0x0" ::: "memory");
    __syncthreads();                               // slab-s data visible to all

    const v16h* As = (const v16h*)(smem + (s & 1) * ABUF_BYTES);
    __builtin_prefetch((const void*)(wfrag +                 // next slab's B
        ((size_t)(s + 1) * 8 + wn * 4) * 32 + lane), 0, 1);

    // ---- software-pipelined K-loop: 9 basis steps x 4 col-tiles ----
    v16h a_cur = As[(mt * NB + 0) * 32 + lane];
    const v16h* wp0 = wfrag + (size_t)s * 8 * 32;            // chunk = b*9+s
    v16h bc0 = wp0[(wn * 4 + 0) * 32 + lane];
    v16h bc1 = wp0[(wn * 4 + 1) * 32 + lane];
    v16h bc2 = wp0[(wn * 4 + 2) * 32 + lane];
    v16h bc3 = wp0[(wn * 4 + 3) * 32 + lane];
#pragma unroll
    for (int b = 0; b < NB; ++b) {
      v16h a_nxt, bn0, bn1, bn2, bn3;
      if (b < NB - 1) {                            // prefetch step b+1 operands
        a_nxt = As[(mt * NB + b + 1) * 32 + lane];
        const v16h* wp = wfrag + (size_t)((b + 1) * 9 + s) * 8 * 32;
        bn0 = wp[(wn * 4 + 0) * 32 + lane];
        bn1 = wp[(wn * 4 + 1) * 32 + lane];
        bn2 = wp[(wn * 4 + 2) * 32 + lane];
        bn3 = wp[(wn * 4 + 3) * 32 + lane];
      }
      acc[0] = __builtin_amdgcn_wmma_f32_16x16x32_f16(false, a_cur, false, bc0, (short)0, acc[0], false, false);
      acc[1] = __builtin_amdgcn_wmma_f32_16x16x32_f16(false, a_cur, false, bc1, (short)0, acc[1], false, false);
      acc[2] = __builtin_amdgcn_wmma_f32_16x16x32_f16(false, a_cur, false, bc2, (short)0, acc[2], false, false);
      acc[3] = __builtin_amdgcn_wmma_f32_16x16x32_f16(false, a_cur, false, bc3, (short)0, acc[3], false, false);
      if (b < NB - 1) { a_cur = a_nxt; bc0 = bn0; bc1 = bn1; bc2 = bn2; bc3 = bn3; }
    }
    __syncthreads();   // all reads of buf[s&1] done before it is re-staged (s+2)
  }

  // ---- epilogue: bias + store (C/D: row = vgpr + 8*(lane/16), col = lane%16)
#pragma unroll
  for (int q = 0; q < 4; ++q) {
    int col = (wn * 4 + q) * 16 + (lane & 15);
    float bv = bias[col];
#pragma unroll
    for (int v = 0; v < 8; ++v) {
      int row = mt * 16 + v + ((lane >> 4) << 3);
      out[(size_t)(tile_base + row) * FOUT + col] = acc[q][v] + bv;
    }
  }
}

// ---------------------------------------------------------------------------
extern "C" void kernel_launch(void* const* d_in, const int* in_sizes, int n_in,
                              void* d_out, int out_size, void* d_ws, size_t ws_size,
                              hipStream_t stream) {
  (void)in_sizes; (void)n_in; (void)out_size; (void)ws_size;
  const float* x  = (const float*)d_in[0];
  const float* sk = (const float*)d_in[1];
  const float* sc = (const float*)d_in[2];
  const float* kb = (const float*)d_in[3];
  const float* cb = (const float*)d_in[4];

  unsigned int* wfrag = (unsigned int*)d_ws;
  float* biasws = (float*)((char*)d_ws + WFRAG_BYTES);
  _Float16* planes = (_Float16*)((char*)d_ws + PLANES_OFF);

  kan_prep <<<WFRAG_U32 / 256, 256, 0, stream>>>(sk, sc, kb, cb, wfrag, biasws);
  kan_basis<<<PLANE_ELEMS / (256 * 8), 256, 0, stream>>>(x, planes);
  kan_gemm <<<NPATCH / MTILE, 256, 2 * ABUF_BYTES, stream>>>(
      (const v16h*)d_ws, planes, biasws, (float*)d_out);
}
